// GraphormerMultiHeadAttention_15779709846381
// MI455X (gfx1250) — matrix operand order repaired and checked by reading, not compiled, not run on hardware
//
#include <hip/hip_runtime.h>

typedef __attribute__((ext_vector_type(16))) __bf16 v16bf;
typedef __attribute__((ext_vector_type(8)))  float  v8f;

union V16 { v16bf v; uint4 u[2]; };

constexpr int B_ = 16, N_ = 512, HID = 768, NH = 12, HD = 64;
constexpr int MTOT = B_ * N_; // 8192
constexpr float SCALE = 0.036084391824351615f; // 768^-0.5

// round-half-up f32 -> bf16 (add + d16_hi store)
__device__ __forceinline__ unsigned short f2bf_fast(float f) {
  return (unsigned short)((__float_as_uint(f) + 0x8000u) >> 16);
}

// two f32 -> packed bf16x2 via v_perm_b32
__device__ __forceinline__ unsigned int pk2bf(float lo, float hi) {
  unsigned int rl = __float_as_uint(lo) + 0x8000u;
  unsigned int rh = __float_as_uint(hi) + 0x8000u;
  return __builtin_amdgcn_perm(rh, rl, 0x07060302u);
}

// exp both bf16 halves of u, accumulate into s, return packed bf16 exps
__device__ __forceinline__ unsigned int exppk_u32(unsigned int u, float mx, float& s) {
  float a = __expf(__uint_as_float(u << 16) - mx);
  float b = __expf(__uint_as_float(u & 0xffff0000u) - mx);
  s += a + b;
  return pk2bf(a, b);
}

__device__ __forceinline__ v8f wmma_bf16(v16bf a, v16bf b, v8f c) {
  return __builtin_amdgcn_wmma_f32_16x16x32_bf16(false, a, false, b, (short)0, c,
                                                 false, false);
}

__global__ void pack_bf16(const float4* __restrict__ in,
                          uint2* __restrict__ out, int n4) {
  int i = blockIdx.x * blockDim.x + threadIdx.x;
  if (i < n4) {
    float4 v = in[i];
    uint2 o;
    o.x = pk2bf(v.x, v.y);
    o.y = pk2bf(v.z, v.w);
    out[i] = o;
  }
}

// ---------------------------------------------------------------------------
// GEMM: Y[8192 x 768] = A(bf16) @ W(bf16)^T, 8 waves/block, block tile 256x64,
// wave tile 32x64. Weight tile staged in LDS, ping-pong buffers (no rotation
// copies), padded stride for conflict-free ds_load_b128.
// MODE 0: store bf16 row-major [token][768]            (Q, K)
// MODE 1: +bias, store bf16 transposed Vt[b][h][d][n]  (V)
// MODE 2: +bias, store f32 row-major                   (final projection)
// ---------------------------------------------------------------------------
constexpr int WPAD = 40;

template <int MODE>
__global__ void __launch_bounds__(256)
gemm16(const unsigned short* __restrict__ A,
       const unsigned short* __restrict__ W,
       const float* __restrict__ bias, void* __restrict__ outp) {
  __shared__ __align__(16) unsigned short wt[2][64 * WPAD]; // 10 KB

  int tid  = threadIdx.x;
  int wid  = tid >> 5;
  int lane = tid & 31;
  int m0 = blockIdx.x * 256 + wid * 32;
  int n0 = blockIdx.y * 64;
  int lrow = lane & 15;
  int lhi  = lane >> 4;
  int s8   = lhi * 8;

  int wr = tid >> 2;
  int wc = (tid & 3) * 8;
  int wofs = wr * WPAD + wc;
  const unsigned short* wsrc = W + (size_t)(n0 + wr) * HID + wc;

  const unsigned short* arow0 = A + (size_t)(m0 + lrow) * HID;
  const unsigned short* arow1 = arow0 + (size_t)16 * HID;

  v8f acc[2][4] = {};

  auto loadA = [&](int k, V16& f0, V16& f1) {
    f0.u[0] = *(const uint4*)(arow0 + k + s8);
    f0.u[1] = *(const uint4*)(arow0 + k + 16 + s8);
    f1.u[0] = *(const uint4*)(arow1 + k + s8);
    f1.u[1] = *(const uint4*)(arow1 + k + 16 + s8);
  };
  auto loadBM = [&](int buf, V16 bm[4]) {
#pragma unroll
    for (int j = 0; j < 4; ++j) {
      const unsigned short* bsrc = &wt[buf][(j * 16 + lrow) * WPAD + lhi * 8];
      bm[j].u[0] = *(const uint4*)(bsrc);
      bm[j].u[1] = *(const uint4*)(bsrc + 16);
    }
  };
  auto domm = [&](const V16& f0, const V16& f1, V16 bm[4]) {
#pragma unroll
    for (int j = 0; j < 4; ++j) {
      acc[0][j] = wmma_bf16(f0.v, bm[j].v, acc[0][j]);
      acc[1][j] = wmma_bf16(f1.v, bm[j].v, acc[1][j]);
    }
  };

  V16 aA0, aA1, aB0, aB1;
  loadA(0, aA0, aA1);
  *(uint4*)&wt[0][wofs] = *(const uint4*)(wsrc);

  for (int k0 = 0; k0 < HID; k0 += 64) {
    __syncthreads();
    { // phase 0: compute k0 from wt[0]/aA; stage k0+32 into wt[1]/aB
      V16 bm[4];
      loadBM(0, bm);
      loadA(k0 + 32, aB0, aB1);                    // k0+32 < 768 always
      uint4 g = *(const uint4*)(wsrc + k0 + 32);
      __builtin_prefetch(arow0 + k0 + 96, 0, 0);
      __builtin_prefetch(arow1 + k0 + 96, 0, 0);
      domm(aA0, aA1, bm);
      *(uint4*)&wt[1][wofs] = g;                   // wait lands after WMMAs
    }
    __syncthreads();
    { // phase 1: compute k0+32 from wt[1]/aB; stage k0+64 into wt[0]/aA
      V16 bm[4];
      loadBM(1, bm);
      bool more = (k0 + 64) < HID;
      uint4 g;
      if (more) {
        loadA(k0 + 64, aA0, aA1);
        g = *(const uint4*)(wsrc + k0 + 64);
      }
      domm(aB0, aB1, bm);
      if (more) *(uint4*)&wt[0][wofs] = g;
    }
  }

#pragma unroll
  for (int mt = 0; mt < 2; ++mt)
#pragma unroll
    for (int j = 0; j < 4; ++j)
#pragma unroll
      for (int i = 0; i < 8; ++i) {
        int r = m0 + mt * 16 + i + lhi * 8;
        int c = n0 + j * 16 + lrow;
        float v = acc[mt][j][i];
        if constexpr (MODE == 0) {
          ((unsigned short*)outp)[(size_t)r * HID + c] = f2bf_fast(v);
        } else if constexpr (MODE == 1) {
          v += bias[c];
          int bb = r >> 9, nloc = r & 511;
          int hh = c >> 6, d = c & 63;
          ((unsigned short*)outp)[(((size_t)(bb * NH + hh) * HD + d) << 9) + nloc] =
              f2bf_fast(v);
        } else {
          ((float*)outp)[(size_t)r * HID + c] = v + bias[c];
        }
      }
}

// ---------------------------------------------------------------------------
// Attention: one wave per (b, h, 16-query block). bf16 score/prob buffer in
// LDS; row max + abs-max (pad mask) in registers; single exp pass writes
// unnormalized probs; 1/sum folded into the O store. Ping-pong K/V buffers.
// ---------------------------------------------------------------------------
constexpr int SPW = 520; // bf16 stride of score/prob rows (bank-padded)

__global__ void __launch_bounds__(32)
attn(const unsigned short* __restrict__ qb,
     const unsigned short* __restrict__ kb,
     const unsigned short* __restrict__ vtb,
     const float* __restrict__ ebias,
     unsigned short* __restrict__ ob) {
  __shared__ __align__(16) unsigned short sp[16 * SPW];
  int lane = threadIdx.x;
  int idx = blockIdx.x;
  int qt = idx & 31;
  int h  = (idx >> 5) % NH;
  int b  = idx / (32 * NH);
  int m0 = qt * 16;
  int lrow = lane & 15;
  int lhi  = lane >> 4;
  int s8   = lhi * 8;

  // Q fragment (16 x 64)
  const unsigned short* qrow = qb + (size_t)(b * N_ + m0 + lrow) * HID + h * HD;
  V16 a0, a1;
  a0.u[0] = *(const uint4*)(qrow + s8);
  a0.u[1] = *(const uint4*)(qrow + 16 + s8);
  a1.u[0] = *(const uint4*)(qrow + 32 + s8);
  a1.u[1] = *(const uint4*)(qrow + 48 + s8);

  const unsigned short* kbase = kb + (size_t)(b * N_) * HID + h * HD + lhi * 8;
  auto loadK = [&](int kt, V16& c0, V16& c1) {
    const unsigned short* krow = kbase + (size_t)(kt * 16 + lrow) * HID;
    c0.u[0] = *(const uint4*)(krow);
    c0.u[1] = *(const uint4*)(krow + 16);
    c1.u[0] = *(const uint4*)(krow + 32);
    c1.u[1] = *(const uint4*)(krow + 48);
  };

  float rmax[8], aabs[8];
#pragma unroll
  for (int i = 0; i < 8; ++i) { rmax[i] = -3.4e38f; aabs[i] = 0.0f; }

  const float* brow0 = ebias + ((size_t)(b * N_) + m0) * N_;
  auto scoreTile = [&](int kt, const V16& f0, const V16& f1) {
    v8f c = {};
    c = wmma_bf16(a0.v, f0.v, c);
    c = wmma_bf16(a1.v, f1.v, c);
    int col = kt * 16 + lrow;
#pragma unroll
    for (int i = 0; i < 8; ++i) {
      int r = i + lhi * 8;
      float sv = c[i] * SCALE + brow0[(size_t)r * N_ + col];
      rmax[i] = fmaxf(rmax[i], sv);
      aabs[i] = fmaxf(aabs[i], __builtin_fabsf(sv)); // |x| is a free modifier
      sp[r * SPW + col] = f2bf_fast(sv);
    }
  };

  // scores -> bf16 LDS (ping-pong K fragments, no rotation copies)
  V16 kA0, kA1, kB0, kB1;
  loadK(0, kA0, kA1);
  for (int kt = 0; kt < 32; kt += 2) {
    loadK(kt + 1, kB0, kB1);
    scoreTile(kt, kA0, kA1);
    if (kt + 2 < 32) loadK(kt + 2, kA0, kA1);
    scoreTile(kt + 1, kB0, kB1);
  }
  // reduce max / abs-max across the 16 lanes sharing each row
#pragma unroll
  for (int m = 1; m < 16; m <<= 1)
#pragma unroll
    for (int i = 0; i < 8; ++i) {
      rmax[i] = fmaxf(rmax[i], __shfl_xor(rmax[i], m, 32));
      aabs[i] = fmaxf(aabs[i], __shfl_xor(aabs[i], m, 32));
    }
  __syncthreads();

  // single pass: exp(s - max), accumulate row sum, write unnormalized probs
  int l15 = lane & 15;
  int ph  = l15 >> 2;              // bank-conflict phase stagger
  float rsum[8];
#pragma unroll
  for (int i = 0; i < 8; ++i) {
    int r = i + lhi * 8;
    uint4* pr = (uint4*)&sp[r * SPW + l15 * 32];
    float s = 0.0f;
#pragma unroll
    for (int q = 0; q < 4; ++q) {
      int qi = (q + ph) & 3;
      uint4 u = pr[qi];
      uint4 w_;
      w_.x = exppk_u32(u.x, rmax[i], s);
      w_.y = exppk_u32(u.y, rmax[i], s);
      w_.z = exppk_u32(u.z, rmax[i], s);
      w_.w = exppk_u32(u.w, rmax[i], s);
      pr[qi] = w_;
    }
    rsum[i] = s;
  }
#pragma unroll
  for (int m = 1; m < 16; m <<= 1)
#pragma unroll
    for (int i = 0; i < 8; ++i) rsum[i] += __shfl_xor(rsum[i], m, 32);
  float inv[8];
#pragma unroll
  for (int i = 0; i < 8; ++i)
    inv[i] = (aabs[i] == 0.0f) ? 0.0f : (1.0f / rsum[i]); // pad row -> 0
  __syncthreads();

  // O = P.V (ping-pong V fragments); normalization applied at the store
  const unsigned short* vbase =
      vtb + (size_t)((b * NH + h) * HD + lrow) * N_ + lhi * 8;
  auto loadV = [&](int kc, V16 vv[4]) {
#pragma unroll
    for (int j = 0; j < 4; ++j) {
      const unsigned short* vrow = vbase + (size_t)(j * 16) * N_ + kc;
      vv[j].u[0] = *(const uint4*)(vrow);
      vv[j].u[1] = *(const uint4*)(vrow + 16);
    }
  };
  v8f o[4] = {};
  auto pvTile = [&](int kc, V16 vv[4]) {
    V16 pa;
    pa.u[0] = *(const uint4*)(&sp[lrow * SPW + kc + s8]);
    pa.u[1] = *(const uint4*)(&sp[lrow * SPW + kc + 16 + s8]);
#pragma unroll
    for (int j = 0; j < 4; ++j) o[j] = wmma_bf16(pa.v, vv[j].v, o[j]);
  };

  V16 vA[4], vB[4];
  loadV(0, vA);
  for (int kc = 0; kc < 512; kc += 64) {
    loadV(kc + 32, vB);
    pvTile(kc, vA);
    if (kc + 64 < 512) loadV(kc + 64, vA);
    pvTile(kc + 32, vB);
  }

#pragma unroll
  for (int j = 0; j < 4; ++j)
#pragma unroll
    for (int i = 0; i < 8; ++i) {
      int r = m0 + i + lhi * 8;
      int c = h * HD + j * 16 + lrow;
      ob[(size_t)(b * N_ + r) * HID + c] = f2bf_fast(o[j][i] * inv[i]);
    }
}

extern "C" void kernel_launch(void* const* d_in, const int* in_sizes, int n_in,
                              void* d_out, int out_size, void* d_ws, size_t ws_size,
                              hipStream_t stream) {
  (void)in_sizes; (void)n_in; (void)out_size; (void)ws_size;
  const float* x     = (const float*)d_in[0];
  const float* ebias = (const float*)d_in[1];
  const float* Wq    = (const float*)d_in[2];
  const float* Wk    = (const float*)d_in[3];
  const float* Wv    = (const float*)d_in[4];
  const float* bv    = (const float*)d_in[5];
  const float* Wo    = (const float*)d_in[6];
  const float* bo    = (const float*)d_in[7];
  float* out = (float*)d_out;

  unsigned short* w    = (unsigned short*)d_ws;
  unsigned short* xb   = w;
  unsigned short* wqb  = xb  + (size_t)MTOT * HID;
  unsigned short* wkb  = wqb + (size_t)HID * HID;
  unsigned short* wvb  = wkb + (size_t)HID * HID;
  unsigned short* wob  = wvb + (size_t)HID * HID;
  unsigned short* qbuf = wob + (size_t)HID * HID;
  unsigned short* kbuf = qbuf + (size_t)MTOT * HID;
  unsigned short* vtb  = kbuf + (size_t)MTOT * HID;
  unsigned short* obuf = vtb  + (size_t)MTOT * HID;

  const int nx4 = MTOT * HID / 4;
  const int nw4 = HID * HID / 4;
  pack_bf16<<<(nx4 + 255) / 256, 256, 0, stream>>>((const float4*)x,  (uint2*)xb,  nx4);
  pack_bf16<<<(nw4 + 255) / 256, 256, 0, stream>>>((const float4*)Wq, (uint2*)wqb, nw4);
  pack_bf16<<<(nw4 + 255) / 256, 256, 0, stream>>>((const float4*)Wk, (uint2*)wkb, nw4);
  pack_bf16<<<(nw4 + 255) / 256, 256, 0, stream>>>((const float4*)Wv, (uint2*)wvb, nw4);
  pack_bf16<<<(nw4 + 255) / 256, 256, 0, stream>>>((const float4*)Wo, (uint2*)wob, nw4);

  dim3 g(MTOT / 256, HID / 64);  // 32 x 12 blocks, 256 threads (8 waves)
  gemm16<0><<<g, 256, 0, stream>>>(xb, wqb, nullptr, qbuf);
  gemm16<0><<<g, 256, 0, stream>>>(xb, wkb, nullptr, kbuf);
  gemm16<1><<<g, 256, 0, stream>>>(xb, wvb, bv,      vtb);

  attn<<<B_ * NH * (N_ / 16), 32, 0, stream>>>(qbuf, kbuf, vtb, ebias, obuf);

  gemm16<2><<<g, 256, 0, stream>>>(obuf, wob, bo, out);
}
